// PonitaFixedSize_50517405335850
// MI455X (gfx1250) — matrix operand order repaired and checked by reading, not compiled, not run on hardware
//
#include <hip/hip_runtime.h>
#include <hip/hip_bf16.h>
#include <math.h>

typedef __attribute__((ext_vector_type(16))) _Float16 v16h;
typedef __attribute__((ext_vector_type(8)))  float    v8f;
typedef int v4i_ __attribute__((vector_size(16)));
typedef __attribute__((address_space(1))) v4i_ gv4i;   // global 16B vector
typedef __attribute__((address_space(3))) v4i_ lv4i;   // LDS 16B vector

#define DEVFN __device__ __forceinline__

constexpr int Npts = 256;
constexpr int Ornt = 12;
constexpr int Hdim = 128;
constexpr int BDim = 64;
constexpr int NIn  = 16;
constexpr int WFH  = 512;   // WF * H
constexpr int NCHUNK = Npts/16;
constexpr float TWO_PI = 6.2831853071795864769f;

// gfx1250 async global->LDS path (ASYNCcnt). Guarded so the file still
// compiles (with a synchronous fallback) if the builtins are absent.
#if defined(__gfx1250__) && __has_builtin(__builtin_amdgcn_global_load_async_to_lds_b128) && __has_builtin(__builtin_amdgcn_s_wait_asynccnt)
#define PONITA_ASYNC_LDS 1
#else
#define PONITA_ASYNC_LDS 0
#endif

DEVFN float gelu_f(float x){
  return 0.5f*x*(1.0f + tanhf(0.7978845608028654f*(x + 0.044715f*x*x*x)));
}

DEVFN v8f wmma_f16(v16h a, v16h b, v8f c){
  // D = A(16x32 f16) * B(32x16 f16) + C(16x16 f32)
  return __builtin_amdgcn_wmma_f32_16x16x32_f16(false, a, false, b, (short)0, c, false, false);
}

// B fragment (32x16) from [N][K] f16 storage (kstride halves per row, 32B-aligned rows).
DEVFN v16h load_b_frag(const _Float16* base, int kstride, int k0, int tile){
  int lane = threadIdx.x & 31;
  const _Float16* p = base + (size_t)(tile*16 + (lane & 15))*kstride + k0 + ((lane >> 4) << 3);
  v16h r;
#pragma unroll
  for (int i=0;i<8;++i) r[i] = p[i];
#pragma unroll
  for (int i=0;i<8;++i) r[8+i] = p[16+i];
  return r;
}

// A fragment (16x32) from [M][K] f16 storage.
DEVFN v16h load_a_frag(const _Float16* base, int kstride, int k0){
  int lane = threadIdx.x & 31;
  int hi = lane >> 4;
  const _Float16* row = base + (size_t)(lane & 15)*kstride + k0 + 8*hi;
  v16h r;
#pragma unroll
  for (int i=0;i<8;++i) r[i]   = row[i];
#pragma unroll
  for (int i=0;i<8;++i) r[8+i] = row[16+i];
  return r;
}

// Stage one 16x128 f32 h-tile (rows n0..n0+15, fixed o) into LDS.
DEVFN void issue_h_stage(const float* h_in, int o, int n0, float* dst, int lane){
#if PONITA_ASYNC_LDS
#pragma unroll
  for (int i=0;i<16;++i){
    const float* g = h_in + ((size_t)(n0+i)*Ornt + o)*Hdim + lane*4;
    float* l = dst + i*Hdim + lane*4;
    __builtin_amdgcn_global_load_async_to_lds_b128((gv4i*)g, (lv4i*)l, 0, 0);
  }
#else
#pragma unroll
  for (int i=0;i<16;++i){
    const float4 v = *(const float4*)(h_in + ((size_t)(n0+i)*Ornt + o)*Hdim + lane*4);
    *(float4*)(dst + i*Hdim + lane*4) = v;
  }
#endif
}

// ---------------------------------------------------------------------------
// Pack fp32 weights -> f16 [N][K] buffers for B-fragment loading.
__global__ void __launch_bounds__(256) ponita_pack_kernel(
    const float* sk_w1, const float* sk_w2, const float* conv_sp_w,
    const float* w1, const float* w2,
    _Float16* gB1, _Float16* gB2, _Float16* gB3, _Float16* gW1, _Float16* gW2)
{
  int idx = blockIdx.x*blockDim.x + threadIdx.x;
  int stride = gridDim.x*blockDim.x;
  // gB1: [128][32], K padded 14->32; src sk_w1 (14,128)
  for (int i=idx; i<128*32; i+=stride){ int c=i>>5, k=i&31;
    gB1[i] = (k<14) ? (_Float16)sk_w1[k*Hdim + c] : (_Float16)0.0f; }
  // gB2: [64][128]; src sk_w2 (128,64)
  for (int i=idx; i<64*128; i+=stride){ int c=i>>7, k=i&127;
    gB2[i] = (_Float16)sk_w2[k*BDim + c]; }
  // gB3: 2 x [128][64]; src conv_sp_w (2,64,128)
  for (int i=idx; i<2*128*64; i+=stride){ int l=i>>13, r=i&8191, c=r>>6, k=r&63;
    gB3[i] = (_Float16)conv_sp_w[((size_t)l*BDim + k)*Hdim + c]; }
  // gW1: 2 x [512][128]; src w1 (2,128,512)
  for (int i=idx; i<2*512*128; i+=stride){ int l=i>>16, r=i&65535, c=r>>7, k=r&127;
    gW1[i] = (_Float16)w1[((size_t)l*Hdim + k)*WFH + c]; }
  // gW2: 2 x [128][512]; src w2 (2,512,128)
  for (int i=idx; i<2*128*512; i+=stride){ int l=i>>16, r=i&65535, c=r>>9, k=r&511;
    gW2[i] = (_Float16)w2[((size_t)l*WFH + k)*Hdim + c]; }
}

// ---------------------------------------------------------------------------
// Orientation kernel basis fb (144x64) and rk[l][p][o][c] = fb @ conv_rot_w[l].
__global__ void __launch_bounds__(128) ponita_small_kernel(
    const float* rk_w1, const float* rk_b1, const float* rk_w2, const float* rk_b2,
    const float* conv_rot_w, float* rk_out)
{
  __shared__ float s_hid[Hdim];
  __shared__ float s_fb[BDim];
  int tid = threadIdx.x;
  for (int pair=0; pair<Ornt*Ornt; ++pair){
    int p = pair/Ornt, o = pair%Ornt;
    float tp = TWO_PI*p/Ornt, to = TWO_PI*o/Ornt;
    float r = cosf(tp)*cosf(to) + sinf(tp)*sinf(to);
    float f0=r, f1=r*r, f2=r*r*r;
    if (tid < Hdim)
      s_hid[tid] = gelu_f(f0*rk_w1[tid] + f1*rk_w1[Hdim+tid] + f2*rk_w1[2*Hdim+tid] + rk_b1[tid]);
    __syncthreads();
    if (tid < BDim){
      float a = rk_b2[tid];
      for (int c=0;c<Hdim;++c) a += s_hid[c]*rk_w2[c*BDim+tid];
      s_fb[tid] = gelu_f(a);
    }
    __syncthreads();
    for (int l=0;l<2;++l){
      if (tid < Hdim){
        float a = 0.f;
        for (int d=0;d<BDim;++d) a += s_fb[d]*conv_rot_w[((size_t)l*BDim+d)*Hdim + tid];
        rk_out[((size_t)l*144 + pair)*Hdim + tid] = a;
      }
    }
    __syncthreads();
  }
}

// ---------------------------------------------------------------------------
// h[n][o][c] = x[n] @ emb_w, replicated over orientations.
__global__ void __launch_bounds__(128) ponita_embed_kernel(
    const float* x, const float* emb_w, float* hbuf)
{
  int n = blockIdx.x, c = threadIdx.x;
  float a = 0.f;
  for (int k=0;k<NIn;++k) a += x[n*NIn+k]*emb_w[k*Hdim+c];
  for (int o=0;o<Ornt;++o) hbuf[((size_t)n*Ornt+o)*Hdim + c] = a;
}

// ---------------------------------------------------------------------------
// Fused spatial conv: for block (m,o) loop n-chunks of 16; recompute kb tile
// via 3 WMMA GEMMs and contract against LDS-staged h[n,o,:] on the fly.
// h tiles are double-buffered via async global->LDS DMA (ASYNCcnt).
__global__ void __launch_bounds__(32) ponita_conv_kernel(
    const float* pos, const float* sk_b1, const float* sk_b2,
    const _Float16* gB1, const _Float16* gB2, const _Float16* gB3l,
    const float* h_in, float* y_out)
{
  __shared__ _Float16 sQ[16*Hdim];
  __shared__ _Float16 sKB[16*BDim];
  __shared__ float    sH[2][16*Hdim];   // double-buffered h tile (2 x 8KB)
  int m = blockIdx.x, o = blockIdx.y;
  int lane = threadIdx.x & 31;
  int li = lane & 15, hi = lane >> 4;
  float th = TWO_PI*o/Ornt;
  float ct = cosf(th), st = sinf(th);
  float pmx = pos[2*m], pmy = pos[2*m+1];
  float yacc[8];
#pragma unroll
  for (int t=0;t<8;++t) yacc[t]=0.f;

  // Loop-invariant B fragments for GEMM1 (8 frags = 64 VGPRs), hoisted.
  v16h b1f[8];
#pragma unroll
  for (int t=0;t<8;++t) b1f[t] = load_b_frag(gB1, 32, 0, t);

  // Prime the h-tile pipeline with chunk 0.
  issue_h_stage(h_in, o, 0, sH[0], lane);

  for (int nc=0; nc<NCHUNK; ++nc){
    int cur = nc & 1;
    if (nc+1 < NCHUNK) issue_h_stage(h_in, o, (nc+1)*16, sH[(nc+1)&1], lane);
#if PONITA_ASYNC_LDS
    if (nc+1 < NCHUNK) __builtin_amdgcn_s_wait_asynccnt(16);  // chunk nc landed
    else               __builtin_amdgcn_s_wait_asynccnt(0);
#endif
    __syncthreads();

    int n = nc*16 + li;
    float rx = pos[2*n]-pmx, ry = pos[2*n+1]-pmy;
    float a = rx*ct + ry*st;
    float b = -rx*st + ry*ct;
    float f[14];
    f[0]=a;    f[1]=b;
    f[2]=a*a;  f[3]=a*b;  f[4]=b*a;  f[5]=b*b;
    f[6]=f[2]*a; f[7]=f[2]*b; f[8]=f[3]*a; f[9]=f[3]*b;
    f[10]=f[4]*a; f[11]=f[4]*b; f[12]=f[5]*a; f[13]=f[5]*b;
    v16h afrag;
#pragma unroll
    for (int hh=0; hh<16; ++hh){
      int Klo = hh + ((hh>=8)?8:0);      // K for lanes 0-15
      int Khi = Klo + 8;                  // K for lanes 16-31
      float flo = (Klo<14) ? f[Klo] : 0.f;
      float fhi = (Khi<14) ? f[Khi] : 0.f;
      afrag[hh] = (_Float16)(hi ? fhi : flo);
    }
    // GEMM1: (16x14pad32) @ sk_w1 -> gelu -> sQ (16x128)
#pragma unroll
    for (int t=0;t<8;++t){
      v8f acc = {};
      acc = wmma_f16(afrag, b1f[t], acc);
      int c = t*16 + li;
      float bias = sk_b1[c];
#pragma unroll
      for (int j=0;j<8;++j)
        sQ[(j + 8*hi)*Hdim + c] = (_Float16)gelu_f(acc[j] + bias);
    }
    __syncthreads();
    // GEMM2: sQ (16x128) @ sk_w2 (128x64) -> gelu -> sKB (16x64)
#pragma unroll
    for (int t=0;t<4;++t){
      v8f acc = {};
#pragma unroll
      for (int kk=0;kk<4;++kk){
        v16h aa = load_a_frag(sQ, Hdim, kk*32);
        v16h bf = load_b_frag(gB2, Hdim, kk*32, t);
        acc = wmma_f16(aa, bf, acc);
      }
      int c = t*16 + li;
      float bias = sk_b2[c];
#pragma unroll
      for (int j=0;j<8;++j)
        sKB[(j + 8*hi)*BDim + c] = (_Float16)gelu_f(acc[j] + bias);
    }
    __syncthreads();
    // GEMM3: sKB (16x64) @ conv_sp_w (64x128) -> sk tile; multiply by LDS h, reduce over n
#pragma unroll
    for (int t=0;t<8;++t){
      v8f acc = {};
#pragma unroll
      for (int kk=0;kk<2;++kk){
        v16h aa = load_a_frag(sKB, BDim, kk*32);
        v16h bf = load_b_frag(gB3l, BDim, kk*32, t);
        acc = wmma_f16(aa, bf, acc);
      }
      int c = t*16 + li;
      float part = 0.f;
#pragma unroll
      for (int j=0;j<8;++j)
        part += acc[j] * sH[cur][(j + 8*hi)*Hdim + c];
      yacc[t] += part;
    }
    __syncthreads();
  }
  // lanes L and L+16 hold partials for the same channel c: combine halves
#pragma unroll
  for (int t=0;t<8;++t){
    float v = yacc[t] + __shfl_xor(yacc[t], 16, 32);
    if (hi == 0) y_out[((size_t)m*Ornt + o)*Hdim + t*16 + li] = v;
  }
}

// ---------------------------------------------------------------------------
// y2[m,p,c] = sum_o y[m,o,c]*rk[p,o,c] + conv_b; then LayerNorm over c -> f16.
__global__ void __launch_bounds__(128) ponita_rotmix_ln_kernel(
    const float* ybuf, const float* rk_l, const float* conv_b_l,
    const float* ln_s_l, const float* ln_b_l, _Float16* ln16)
{
  __shared__ float red[Hdim];
  __shared__ float stat[2];
  int m = blockIdx.x, p = blockIdx.y, c = threadIdx.x;
  float s = conv_b_l[c];
  for (int o=0;o<Ornt;++o)
    s += ybuf[((size_t)m*Ornt+o)*Hdim + c] * rk_l[((size_t)p*Ornt+o)*Hdim + c];
  red[c] = s; __syncthreads();
  for (int off=64; off>0; off>>=1){ if (c<off) red[c]+=red[c+off]; __syncthreads(); }
  if (c==0) stat[0] = red[0]*(1.0f/Hdim);
  __syncthreads();
  float mu = stat[0];
  float d = s - mu;
  red[c] = d*d; __syncthreads();
  for (int off=64; off>0; off>>=1){ if (c<off) red[c]+=red[c+off]; __syncthreads(); }
  if (c==0) stat[1] = red[0]*(1.0f/Hdim);
  __syncthreads();
  float var = stat[1];
  float vo = d*rsqrtf(var + 1e-6f)*ln_s_l[c] + ln_b_l[c];
  ln16[((size_t)m*Ornt+p)*Hdim + c] = (_Float16)vo;
}

// ---------------------------------------------------------------------------
// MLP + residual: h += gelu(ln @ w1 + b1) @ w2 + b2. 16-row tiles, 4 waves.
__global__ void __launch_bounds__(128) ponita_mlp_kernel(
    const _Float16* ln16, const _Float16* gW1l, const _Float16* gW2l,
    const float* b1l, const float* b2l, float* hbuf)
{
  __shared__ _Float16 sH1[16*WFH];
  int row0 = blockIdx.x*16;
  int w  = threadIdx.x >> 5;
  int lane = threadIdx.x & 31;
  int li = lane & 15, hi = lane >> 4;
  const _Float16* Abase = ln16 + (size_t)row0*Hdim;
  // GEMM1: (16x128)@(128x512) -> gelu -> sH1; wave w owns c-tiles [8w, 8w+8)
  for (int t = w*8; t < w*8+8; ++t){
    v8f acc = {};
#pragma unroll
    for (int kk=0;kk<4;++kk){
      v16h aa = load_a_frag(Abase, Hdim, kk*32);
      v16h bf = load_b_frag(gW1l, Hdim, kk*32, t);
      acc = wmma_f16(aa, bf, acc);
    }
    int c = t*16 + li;
    float bias = b1l[c];
#pragma unroll
    for (int j=0;j<8;++j)
      sH1[(j + 8*hi)*WFH + c] = (_Float16)gelu_f(acc[j] + bias);
  }
  __syncthreads();
  // GEMM2: (16x512)@(512x128); wave w owns 2 c-tiles; residual add into hbuf
  for (int t2 = w*2; t2 < w*2+2; ++t2){
    v8f acc = {};
#pragma unroll
    for (int kk=0;kk<16;++kk){
      v16h aa = load_a_frag(sH1, WFH, kk*32);
      v16h bf = load_b_frag(gW2l, WFH, kk*32, t2);
      acc = wmma_f16(aa, bf, acc);
    }
    int c = t2*16 + li;
    float bias = b2l[c];
#pragma unroll
    for (int j=0;j<8;++j){
      int row = row0 + j + 8*hi;
      hbuf[(size_t)row*Hdim + c] += acc[j] + bias;
    }
  }
}

// ---------------------------------------------------------------------------
// Readout: r = h @ ro_w + ro_b; out_s = mean over o,n; out_v via ori, /O /N.
__global__ void __launch_bounds__(256) ponita_readout_kernel(
    const float* hbuf, const float* ro_w, const float* ro_b, float* out)
{
  __shared__ float r0[Npts], r1[Npts], r2[Npts];
  int n = threadIdx.x;
  float ssum=0.f, v0=0.f, v1=0.f;
  for (int o=0;o<Ornt;++o){
    float rs = ro_b[0], rv = ro_b[1];
    const float* hp = hbuf + ((size_t)n*Ornt + o)*Hdim;
    for (int c=0;c<Hdim;++c){ float hv = hp[c]; rs += hv*ro_w[c*2+0]; rv += hv*ro_w[c*2+1]; }
    float th = TWO_PI*o/Ornt;
    ssum += rs; v0 += rv*cosf(th); v1 += rv*sinf(th);
  }
  r0[n] = ssum*(1.f/Ornt); r1[n] = v0*(1.f/Ornt); r2[n] = v1*(1.f/Ornt);
  __syncthreads();
  for (int off=128; off>0; off>>=1){
    if (n<off){ r0[n]+=r0[n+off]; r1[n]+=r1[n+off]; r2[n]+=r2[n+off]; }
    __syncthreads();
  }
  if (n==0){ out[0]=r0[0]*(1.f/Npts); out[1]=r1[0]*(1.f/Npts); out[2]=r2[0]*(1.f/Npts); }
}

// ---------------------------------------------------------------------------
extern "C" void kernel_launch(void* const* d_in, const int* in_sizes, int n_in,
                              void* d_out, int out_size, void* d_ws, size_t ws_size,
                              hipStream_t stream)
{
  (void)in_sizes; (void)n_in; (void)out_size; (void)ws_size;
  const float* pos   = (const float*)d_in[0];
  const float* x     = (const float*)d_in[1];
  const float* sk_w1 = (const float*)d_in[2];
  const float* sk_b1 = (const float*)d_in[3];
  const float* sk_w2 = (const float*)d_in[4];
  const float* sk_b2 = (const float*)d_in[5];
  const float* rk_w1 = (const float*)d_in[6];
  const float* rk_b1 = (const float*)d_in[7];
  const float* rk_w2 = (const float*)d_in[8];
  const float* rk_b2 = (const float*)d_in[9];
  const float* emb_w = (const float*)d_in[10];
  const float* conv_sp_w  = (const float*)d_in[11];
  const float* conv_rot_w = (const float*)d_in[12];
  const float* conv_b = (const float*)d_in[13];
  const float* ln_s  = (const float*)d_in[14];
  const float* ln_b  = (const float*)d_in[15];
  const float* w1    = (const float*)d_in[16];
  const float* b1    = (const float*)d_in[17];
  const float* w2    = (const float*)d_in[18];
  const float* b2    = (const float*)d_in[19];
  const float* ro_w  = (const float*)d_in[20];
  const float* ro_b  = (const float*)d_in[21];

  char* ws = (char*)d_ws;
  size_t off = 0;
  auto take = [&](size_t bytes)->char*{
    char* p = ws + off; off = (off + bytes + 255) & ~(size_t)255; return p; };
  float*    rk_buf = (float*)   take((size_t)2*144*Hdim*sizeof(float));
  float*    hbuf   = (float*)   take((size_t)Npts*Ornt*Hdim*sizeof(float));
  float*    ybuf   = (float*)   take((size_t)Npts*Ornt*Hdim*sizeof(float));
  _Float16* ln16   = (_Float16*)take((size_t)Npts*Ornt*Hdim*sizeof(_Float16));
  _Float16* gB1    = (_Float16*)take((size_t)128*32*sizeof(_Float16));
  _Float16* gB2    = (_Float16*)take((size_t)64*128*sizeof(_Float16));
  _Float16* gB3    = (_Float16*)take((size_t)2*128*64*sizeof(_Float16));
  _Float16* gW1    = (_Float16*)take((size_t)2*512*128*sizeof(_Float16));
  _Float16* gW2    = (_Float16*)take((size_t)2*128*512*sizeof(_Float16));

  ponita_pack_kernel<<<128, 256, 0, stream>>>(sk_w1, sk_w2, conv_sp_w, w1, w2,
                                              gB1, gB2, gB3, gW1, gW2);
  ponita_small_kernel<<<1, 128, 0, stream>>>(rk_w1, rk_b1, rk_w2, rk_b2, conv_rot_w, rk_buf);
  ponita_embed_kernel<<<Npts, 128, 0, stream>>>(x, emb_w, hbuf);
  for (int l=0; l<2; ++l){
    ponita_conv_kernel<<<dim3(Npts, Ornt), 32, 0, stream>>>(
        pos, sk_b1, sk_b2, gB1, gB2, gB3 + (size_t)l*128*64, hbuf, ybuf);
    ponita_rotmix_ln_kernel<<<dim3(Npts, Ornt), 128, 0, stream>>>(
        ybuf, rk_buf + (size_t)l*144*Hdim, conv_b + l*Hdim,
        ln_s + l*Hdim, ln_b + l*Hdim, ln16);
    ponita_mlp_kernel<<<(Npts*Ornt)/16, 128, 0, stream>>>(
        ln16, gW1 + (size_t)l*512*128, gW2 + (size_t)l*128*512,
        b1 + l*WFH, b2 + l*Hdim, hbuf);
  }
  ponita_readout_kernel<<<1, Npts, 0, stream>>>(hbuf, ro_w, ro_b, (float*)d_out);
}
